// SlidingAttention_71631464563017
// MI455X (gfx1250) — matrix operand order repaired and checked by reading, not compiled, not run on hardware
//
#include <hip/hip_runtime.h>
#include <math.h>

// ---------------------------------------------------------------------------
// Types for CDNA5 WMMA (wave32): v_wmma_f32_16x16x32_f16
// ---------------------------------------------------------------------------
typedef __attribute__((ext_vector_type(16))) _Float16 v16h;
typedef __attribute__((ext_vector_type(8)))  _Float16 v8h;
typedef __attribute__((ext_vector_type(4)))  _Float16 v4h;
typedef __attribute__((ext_vector_type(8)))  float    v8f;

__device__ __forceinline__ v16h pack16(v8h lo, v8h hi) {
  v16h r;
#pragma unroll
  for (int i = 0; i < 8; ++i) { r[i] = lo[i]; r[i + 8] = hi[i]; }
  return r;
}

__device__ __forceinline__ v8f wmma_f16(v16h a, v16h b, v8f c) {
  // (neg_a, A, neg_b, B, c_mod, C, reuse_a, reuse_b)
  return __builtin_amdgcn_wmma_f32_16x16x32_f16(false, a, false, b, (short)0, c,
                                                false, false);
}

// ---------------------------------------------------------------------------
// Problem constants
// ---------------------------------------------------------------------------
#define C_IN     16
#define D_MODEL  512
#define SEQ      2048
#define BATCH    32
#define KERN     256
#define STRIDE   128
#define NWIN     15        // (2048-256)/128 + 1
#define NHEAD    8
#define HD       64

// ---------------------------------------------------------------------------
// Kernel 1: conv1d (wrap pad) + sinusoidal positional encoding -> emb f32
// ---------------------------------------------------------------------------
__global__ __launch_bounds__(256) void conv_pe_kernel(
    const float* __restrict__ path, const float* __restrict__ cw,
    float* __restrict__ emb) {
  long long idx = (long long)blockIdx.x * 256 + threadIdx.x; // < 32*2048*512
  int d = (int)(idx & 511);
  long long bt = idx >> 9;
  int t = (int)(bt & 2047);
  int b = (int)(bt >> 11);
  float acc = 0.f;
#pragma unroll
  for (int k = 0; k < 3; ++k) {
    int tt = (t - 1 + k + SEQ) & (SEQ - 1);
    const float* pr = path + ((long long)b * SEQ + tt) * C_IN;
#pragma unroll
    for (int c = 0; c < C_IN; ++c)
      acc += pr[c] * cw[(k * C_IN + c) * D_MODEL + d];
  }
  int de = d & ~1;
  float div = __expf(-(float)de * (9.210340371976184f / (float)D_MODEL));
  float ang = (float)t * div;
  acc += (d & 1) ? cosf(ang) : sinf(ang);
  emb[idx] = acc;
}

// ---------------------------------------------------------------------------
// Generic batched WMMA GEMM:  Out[g] = act( X[g](MxK) * B(KxN) + bias + Res[g] )
//   - X: f32 or f16, base = (g/gdiv)*xs1 + (g%gdiv)*xs2   (element offsets)
//   - B: "wT" form  (B[k][n] = Bsrc[n*b_ld + k])  or "kn" form (B[k][n]=Bsrc[k*b_ld+n])
//   - M = gridDim.y*128, N = gridDim.z*128, K % 32 == 0
// All behavioral flags are compile-time template params -> 6 straight-line
// specializations, no per-iteration flag branching in the hot loop.
// ---------------------------------------------------------------------------
struct GemmP {
  const void* X; const void* B; const float* bias; const float* res; void* out;
  long long xs1, xs2, bs1, bs2, rs1, rs2, os1, os2;
  int x_ld, b_ld, r_ld, o_ld;
  int K, gdiv;
};

#define BM 128
#define BN 128
#define BKK 32

template <bool XF16, bool BF16, bool BKN, bool RELU, bool OF16, bool BIAS,
          bool RES>
__global__ __launch_bounds__(256) void gemm_wmma_kernel(GemmP p) {
  __shared__ __align__(16) _Float16 lds_a[BM * BKK];
  __shared__ __align__(16) _Float16 lds_b[BN * BKK];

  const int tid = threadIdx.x;
  const int g = blockIdx.x, mb = blockIdx.y, nb = blockIdx.z;
  const int g1 = g / p.gdiv, g2 = g % p.gdiv;
  const int m0b = mb * BM, n0b = nb * BN;
  const long long xbase = (long long)g1 * p.xs1 + (long long)g2 * p.xs2;
  const long long bbase = (long long)g1 * p.bs1 + (long long)g2 * p.bs2;

  const int wid = tid >> 5, lane = tid & 31;
  const int lm = lane & 15, lh = lane >> 4;

  v8f c[8];
#pragma unroll
  for (int j = 0; j < 8; ++j)
#pragma unroll
    for (int r = 0; r < 8; ++r) c[j][r] = 0.f;

  for (int k0 = 0; k0 < p.K; k0 += BKK) {
    // ---- prefetch next k-tile (global_prefetch_b8 path) ----
    if (k0 + BKK < p.K) {
      {
        int r = (tid >> 3);
        int cc = (tid & 7) * 4;
        long long src = xbase + (long long)(m0b + r) * p.x_ld + (k0 + BKK + cc);
        const char* pa = XF16 ? (const char*)((const _Float16*)p.X + src)
                              : (const char*)((const float*)p.X + src);
        __builtin_prefetch(pa, 0, 1);
      }
      {
        long long src;
        if (!BKN) {
          int r = (tid >> 3);
          int cc = (tid & 7) * 4;
          src = bbase + (long long)(n0b + r) * p.b_ld + (k0 + BKK + cc);
        } else {
          int kk = (tid >> 5);
          int n4 = (tid & 31) * 4;
          src = bbase + (long long)(k0 + BKK + kk) * p.b_ld + (n0b + n4);
        }
        const char* pb = BF16 ? (const char*)((const _Float16*)p.B + src)
                              : (const char*)((const float*)p.B + src);
        __builtin_prefetch(pb, 0, 1);
      }
    }
    __syncthreads();
    // ---- stage A tile (BM x BKK), convert to f16 ----
#pragma unroll
    for (int it = 0; it < 4; ++it) {
      int r = it * 32 + (tid >> 3);
      int cc = (tid & 7) * 4;
      long long src = xbase + (long long)(m0b + r) * p.x_ld + (k0 + cc);
      v4h val;
      if constexpr (XF16) {
        val = *(const v4h*)((const _Float16*)p.X + src);
      } else {
        const float4 f = *(const float4*)((const float*)p.X + src);
        v4h t = { (_Float16)f.x, (_Float16)f.y, (_Float16)f.z, (_Float16)f.w };
        val = t;
      }
      *(v4h*)&lds_a[r * BKK + cc] = val;
    }
    // ---- stage B tile into [n][k] layout ----
    if constexpr (!BKN) {
#pragma unroll
      for (int it = 0; it < 4; ++it) {
        int r = it * 32 + (tid >> 3);              // n-local
        int cc = (tid & 7) * 4;                    // k-local
        long long src = bbase + (long long)(n0b + r) * p.b_ld + (k0 + cc);
        v4h val;
        if constexpr (BF16) {
          val = *(const v4h*)((const _Float16*)p.B + src);
        } else {
          const float4 f = *(const float4*)((const float*)p.B + src);
          v4h t = { (_Float16)f.x, (_Float16)f.y, (_Float16)f.z, (_Float16)f.w };
          val = t;
        }
        *(v4h*)&lds_b[r * BKK + cc] = val;
      }
    } else {
#pragma unroll
      for (int it = 0; it < 4; ++it) {
        int kk = it * 8 + (tid >> 5);              // k-local
        int n4 = (tid & 31) * 4;                   // n-local
        long long src = bbase + (long long)(k0 + kk) * p.b_ld + (n0b + n4);
        _Float16 e0, e1, e2, e3;
        if constexpr (BF16) {
          const _Float16* bp = (const _Float16*)p.B;
          e0 = bp[src]; e1 = bp[src + 1]; e2 = bp[src + 2]; e3 = bp[src + 3];
        } else {
          const float* bp = (const float*)p.B;
          e0 = (_Float16)bp[src]; e1 = (_Float16)bp[src + 1];
          e2 = (_Float16)bp[src + 2]; e3 = (_Float16)bp[src + 3];
        }
        lds_b[(n4 + 0) * BKK + kk] = e0;
        lds_b[(n4 + 1) * BKK + kk] = e1;
        lds_b[(n4 + 2) * BKK + kk] = e2;
        lds_b[(n4 + 3) * BKK + kk] = e3;
      }
    }
    __syncthreads();
    // ---- compute: each wave owns a 16 x 128 strip ----
    const int am = wid * 16 + lm;
    const int kb = lh * 8;        // A-frag K chunk base (ISA 16-bit A layout)
    const int kb2 = lh * 16;      // unused; B frags load with kb (same k order as A)
    v8h alo = *(const v8h*)&lds_a[am * BKK + kb];
    v8h ahi = *(const v8h*)&lds_a[am * BKK + 16 + kb];
    v16h a = pack16(alo, ahi);
#pragma unroll
    for (int jn = 0; jn < 8; ++jn) {
      int bn = jn * 16 + lm;
      v8h blo = *(const v8h*)&lds_b[bn * BKK + kb];
      v8h bhi = *(const v8h*)&lds_b[bn * BKK + 16 + kb];
      v16h b = pack16(blo, bhi);
      c[jn] = wmma_f16(a, b, c[jn]);
    }
  }

  // ---- epilogue: bias / residual / relu / store ----
  const long long obase = (long long)g1 * p.os1 + (long long)g2 * p.os2;
  const long long rbase = (long long)g1 * p.rs1 + (long long)g2 * p.rs2;
#pragma unroll
  for (int jn = 0; jn < 8; ++jn) {
#pragma unroll
    for (int r = 0; r < 8; ++r) {
      int row = m0b + wid * 16 + lh * 8 + r;
      int col = n0b + jn * 16 + lm;
      float v = c[jn][r];
      if constexpr (BIAS) v += p.bias[col];
      if constexpr (RES)  v += p.res[rbase + (long long)row * p.r_ld + col];
      if constexpr (RELU) v = v < 0.f ? 0.f : v;
      long long oi = obase + (long long)row * p.o_ld + col;
      if constexpr (OF16) ((_Float16*)p.out)[oi] = (_Float16)v;
      else                ((float*)p.out)[oi] = v;
    }
  }
}

// Specializations used:              XF16  BF16  BKN   RELU  OF16  BIAS  RES
#define GEMM_XW_BIAS_F16OUT  gemm_wmma_kernel<false,false,false,false,true ,true ,false>
#define GEMM_XW_BIAS_RES     gemm_wmma_kernel<false,false,false,false,false,true ,true >
#define GEMM_XW_BIAS_RELU    gemm_wmma_kernel<false,false,false,true ,false,true ,false>
#define GEMM_H_XW_BIAS_RES   gemm_wmma_kernel<true ,false,false,false,false,true ,true >
#define GEMM_SCORES          gemm_wmma_kernel<true ,true ,false,false,false,false,false>
#define GEMM_PV              gemm_wmma_kernel<true ,true ,true ,false,false,false,false>

// ---------------------------------------------------------------------------
// Kernel 3: fused per-(window, head) self-attention for stage B.
//   qkv: f16 (480, 256, 1536)  [q | k | v], head h at column h*64
//   obuf: f16 (480, 256, 512)  heads re-interleaved
// Scores kept in registers (16 v8f / wave); softmax via cross-lane shuffles;
// probs go through LDS to convert C-layout -> A-fragment layout; P @ V WMMA.
// ---------------------------------------------------------------------------
__global__ __launch_bounds__(256) void attn_head_kernel(
    const _Float16* __restrict__ qkv, _Float16* __restrict__ obuf) {
  __shared__ __align__(16) _Float16 lds_k[KERN * HD];     // 32 KB
  __shared__ __align__(16) _Float16 lds_vT[HD * KERN];    // 32 KB (V transposed)
  __shared__ __align__(16) _Float16 lds_p[8 * 16 * KERN]; // 64 KB probs

  const int g = blockIdx.x;
  const int w = g >> 3, h = g & 7;
  const int tid = threadIdx.x;
  const int wid = tid >> 5, lane = tid & 31;
  const int lm = lane & 15, lh = lane >> 4;
  const long long basew = (long long)w * KERN * 1536 + h * HD;

  // ---- stage K (row major) and V (transposed) into LDS ----
#pragma unroll
  for (int it = 0; it < 16; ++it) {
    int r = it * 16 + (tid >> 4);
    int cc = (tid & 15) * 4;
    const _Float16* src = qkv + basew + (long long)r * 1536;
    v4h kv = *(const v4h*)(src + 512 + cc);
    *(v4h*)&lds_k[r * HD + cc] = kv;
    v4h vv = *(const v4h*)(src + 1024 + cc);
    lds_vT[(cc + 0) * KERN + r] = vv[0];
    lds_vT[(cc + 1) * KERN + r] = vv[1];
    lds_vT[(cc + 2) * KERN + r] = vv[2];
    lds_vT[(cc + 3) * KERN + r] = vv[3];
  }
  __syncthreads();

  const float scale = 0.125f;   // 1/sqrt(64)
  const int kb = lh * 8, kb2 = lh * 16;

  for (int pass = 0; pass < 2; ++pass) {
    const int m0 = (pass * 8 + wid) * 16;
    // q fragments straight from global (contiguous 16B per lane)
    const _Float16* qsrc = qkv + basew + (long long)(m0 + lm) * 1536;
    v16h aq[2];
#pragma unroll
    for (int ks = 0; ks < 2; ++ks) {
      v8h lo = *(const v8h*)(qsrc + ks * 32 + kb);
      v8h hi = *(const v8h*)(qsrc + ks * 32 + 16 + kb);
      aq[ks] = pack16(lo, hi);
    }
    // ---- scores S = q k^T  (16 x 256 per wave, in registers) ----
    v8f s[16];
#pragma unroll
    for (int j = 0; j < 16; ++j)
#pragma unroll
      for (int r = 0; r < 8; ++r) s[j][r] = 0.f;
#pragma unroll
    for (int jn = 0; jn < 16; ++jn) {
      int n = jn * 16 + lm;
#pragma unroll
      for (int ks = 0; ks < 2; ++ks) {
        v8h blo = *(const v8h*)&lds_k[n * HD + ks * 32 + kb];
        v8h bhi = *(const v8h*)&lds_k[n * HD + ks * 32 + 16 + kb];
        v16h b = pack16(blo, bhi);
        s[jn] = wmma_f16(aq[ks], b, s[jn]);
      }
    }
    // ---- row softmax (rows live across 16-lane halves in C layout) ----
    float inv_[8];
#pragma unroll
    for (int r = 0; r < 8; ++r) {
      float mx = -3.0e38f;
#pragma unroll
      for (int jn = 0; jn < 16; ++jn) mx = fmaxf(mx, s[jn][r]);
#pragma unroll
      for (int m = 1; m < 16; m <<= 1) mx = fmaxf(mx, __shfl_xor(mx, m, 32));
      float sum = 0.f;
#pragma unroll
      for (int jn = 0; jn < 16; ++jn) {
        float e = __expf((s[jn][r] - mx) * scale);
        s[jn][r] = e;
        sum += e;
      }
#pragma unroll
      for (int m = 1; m < 16; m <<= 1) sum += __shfl_xor(sum, m, 32);
      inv_[r] = 1.f / sum;
    }
    // store normalized probs (f16) to this wave's LDS region
    _Float16* pp = &lds_p[wid * 16 * KERN];
#pragma unroll
    for (int jn = 0; jn < 16; ++jn)
#pragma unroll
      for (int r = 0; r < 8; ++r)
        pp[(lh * 8 + r) * KERN + jn * 16 + lm] = (_Float16)(s[jn][r] * inv_[r]);
    __syncthreads();
    // ---- O = P @ V  (16 x 64 per wave) ----
    v8f o[4];
#pragma unroll
    for (int j = 0; j < 4; ++j)
#pragma unroll
      for (int r = 0; r < 8; ++r) o[j][r] = 0.f;
#pragma unroll
    for (int ks = 0; ks < 8; ++ks) {
      v8h alo = *(const v8h*)&pp[lm * KERN + ks * 32 + kb];
      v8h ahi = *(const v8h*)&pp[lm * KERN + ks * 32 + 16 + kb];
      v16h a = pack16(alo, ahi);
#pragma unroll
      for (int jn = 0; jn < 4; ++jn) {
        int n = jn * 16 + lm;
        v8h blo = *(const v8h*)&lds_vT[n * KERN + ks * 32 + kb];
        v8h bhi = *(const v8h*)&lds_vT[n * KERN + ks * 32 + 16 + kb];
        v16h b = pack16(blo, bhi);
        o[jn] = wmma_f16(a, b, o[jn]);
      }
    }
#pragma unroll
    for (int jn = 0; jn < 4; ++jn)
#pragma unroll
      for (int r = 0; r < 8; ++r) {
        int row = m0 + lh * 8 + r;
        int col = jn * 16 + lm;
        obuf[((long long)w * KERN + row) * D_MODEL + h * HD + col] =
            (_Float16)o[jn][r];
      }
    __syncthreads();
  }
}

// ---------------------------------------------------------------------------
// Kernel 4: row softmax for the scan (rows of 256, scale applied pre-exp).
// One wave per row, 8 rows per block.
// ---------------------------------------------------------------------------
__global__ __launch_bounds__(256) void softmax256_kernel(
    const float* __restrict__ S, _Float16* __restrict__ P, float scale) {
  int row = blockIdx.x * 8 + (threadIdx.x >> 5);
  int lane = threadIdx.x & 31;
  const float* sp = S + (long long)row * 256;
  float v[8];
#pragma unroll
  for (int j = 0; j < 8; ++j) v[j] = sp[j * 32 + lane];
  float mx = -3.0e38f;
#pragma unroll
  for (int j = 0; j < 8; ++j) mx = fmaxf(mx, v[j]);
#pragma unroll
  for (int m = 1; m < 32; m <<= 1) mx = fmaxf(mx, __shfl_xor(mx, m, 32));
  float sum = 0.f;
#pragma unroll
  for (int j = 0; j < 8; ++j) { v[j] = __expf((v[j] - mx) * scale); sum += v[j]; }
#pragma unroll
  for (int m = 1; m < 32; m <<= 1) sum += __shfl_xor(sum, m, 32);
  float inv = 1.f / sum;
  _Float16* pp = P + (long long)row * 256;
#pragma unroll
  for (int j = 0; j < 8; ++j) pp[j * 32 + lane] = (_Float16)(v[j] * inv);
}

// ---------------------------------------------------------------------------
// Host orchestration
// ---------------------------------------------------------------------------
extern "C" void kernel_launch(void* const* d_in, const int* in_sizes, int n_in,
                              void* d_out, int out_size, void* d_ws,
                              size_t ws_size, hipStream_t stream) {
  (void)in_sizes; (void)n_in; (void)out_size; (void)ws_size;
  const float* path       = (const float*)d_in[0];
  const float* conv_w     = (const float*)d_in[1];
  const float* slide_in_w = (const float*)d_in[2];
  const float* slide_in_b = (const float*)d_in[3];
  const float* slide_out_w= (const float*)d_in[4];
  const float* slide_out_b= (const float*)d_in[5];
  const float* lin_w      = (const float*)d_in[6];
  const float* lin_b      = (const float*)d_in[7];
  const float* acc_in_w   = (const float*)d_in[8];
  const float* acc_in_b   = (const float*)d_in[9];
  const float* acc_out_w  = (const float*)d_in[10];
  const float* acc_out_b  = (const float*)d_in[11];

  char* base = (char*)d_ws;
  size_t off = 0;
  auto take = [&](size_t bytes) -> char* {
    char* p = base + off;
    off += (bytes + 255) & ~(size_t)255;
    return p;
  };

  const long long ES = (long long)KERN * D_MODEL;       // 256*512 elems
  const long long QS = (long long)KERN * 3 * D_MODEL;   // 256*1536 elems

  float*    emb   = (float*)take((size_t)BATCH * SEQ * D_MODEL * 4);      // 128 MB
  _Float16* qkv   = (_Float16*)take((size_t)BATCH * NWIN * QS * 2);       // 360 MB
  _Float16* obuf  = (_Float16*)take((size_t)BATCH * NWIN * ES * 2);       // 120 MB
  float*    xs    = (float*)take((size_t)BATCH * NWIN * ES * 4);          // 240 MB
  _Float16* qb    = (_Float16*)take((size_t)BATCH * ES * 2);
  _Float16* kb_   = (_Float16*)take((size_t)BATCH * ES * 2);
  _Float16* vb_   = (_Float16*)take((size_t)BATCH * ES * 2);
  float*    sbuf  = (float*)take((size_t)BATCH * KERN * KERN * 4);
  _Float16* pbuf  = (_Float16*)take((size_t)BATCH * KERN * KERN * 2);
  float*    obuf2 = (float*)take((size_t)BATCH * ES * 4);
  float*    cumA  = (float*)take((size_t)BATCH * ES * 4);
  float*    cumB  = (float*)take((size_t)BATCH * ES * 4);
  float*    attnres = (float*)qkv;  // alias: qkv dead after attn_head_kernel

  // ---- 1. conv + PE ----
  conv_pe_kernel<<<(BATCH * SEQ * D_MODEL) / 256, 256, 0, stream>>>(path, conv_w, emb);

  // ---- 2. QKV GEMM over overlapping windows (g = b*15 + w) ----
  {
    GemmP p{};
    p.X = emb; p.xs1 = (long long)SEQ * D_MODEL; p.xs2 = (long long)STRIDE * D_MODEL; p.x_ld = D_MODEL;
    p.B = slide_in_w; p.bs1 = 0; p.bs2 = 0; p.b_ld = D_MODEL;
    p.bias = slide_in_b; p.res = nullptr;
    p.out = qkv; p.os1 = NWIN * QS; p.os2 = QS; p.o_ld = 3 * D_MODEL;
    p.K = D_MODEL; p.gdiv = NWIN;
    GEMM_XW_BIAS_F16OUT<<<dim3(BATCH * NWIN, 2, 12), 256, 0, stream>>>(p);
  }

  // ---- 3. fused per-(window, head) attention ----
  attn_head_kernel<<<BATCH * NWIN * NHEAD, 256, 0, stream>>>(qkv, obuf);

  // ---- 4. output projection + residual (residual = emb windows) ----
  {
    GemmP p{};
    p.X = obuf; p.xs1 = NWIN * ES; p.xs2 = ES; p.x_ld = D_MODEL;
    p.B = slide_out_w; p.bs1 = 0; p.bs2 = 0; p.b_ld = D_MODEL;
    p.bias = slide_out_b;
    p.res = emb; p.rs1 = (long long)SEQ * D_MODEL; p.rs2 = (long long)STRIDE * D_MODEL; p.r_ld = D_MODEL;
    p.out = attnres; p.os1 = NWIN * ES; p.os2 = ES; p.o_ld = D_MODEL;
    p.K = D_MODEL; p.gdiv = NWIN;
    GEMM_H_XW_BIAS_RES<<<dim3(BATCH * NWIN, 2, 4), 256, 0, stream>>>(p);
  }

  // ---- 5. linear + ReLU -> xs ----
  {
    GemmP p{};
    p.X = attnres; p.xs1 = ES; p.xs2 = 0; p.x_ld = D_MODEL;
    p.B = lin_w; p.bs1 = 0; p.bs2 = 0; p.b_ld = D_MODEL;
    p.bias = lin_b; p.res = nullptr;
    p.out = xs; p.os1 = ES; p.os2 = 0; p.o_ld = D_MODEL;
    p.K = D_MODEL; p.gdiv = 1;
    GEMM_XW_BIAS_RELU<<<dim3(BATCH * NWIN, 2, 4), 256, 0, stream>>>(p);
  }

  // ---- 6. sequential accumulation scan: cum = xs[:,14]; wi = 13..0 ----
  const float scan_scale = 0.044194173824159216f;  // 1/sqrt(512)
  float* cumprev = xs + 14 * ES;     // xs[b,14] with batch stride 15*ES
  long long cum_s1 = (long long)NWIN * ES;
  for (int s = 0; s < 14; ++s) {
    int wi = 13 - s;
    // q' = xs[:,wi] @ Wq^T + bq
    {
      GemmP p{};
      p.X = xs + (long long)wi * ES; p.xs1 = (long long)NWIN * ES; p.xs2 = 0; p.x_ld = D_MODEL;
      p.B = acc_in_w; p.bs1 = 0; p.bs2 = 0; p.b_ld = D_MODEL;
      p.bias = acc_in_b; p.res = nullptr;
      p.out = qb; p.os1 = ES; p.os2 = 0; p.o_ld = D_MODEL;
      p.K = D_MODEL; p.gdiv = 1;
      GEMM_XW_BIAS_F16OUT<<<dim3(BATCH, 2, 4), 256, 0, stream>>>(p);
    }
    // k' = cum @ Wk^T + bk ; v' = cum @ Wv^T + bv
    for (int kv = 0; kv < 2; ++kv) {
      GemmP p{};
      p.X = cumprev; p.xs1 = cum_s1; p.xs2 = 0; p.x_ld = D_MODEL;
      p.B = acc_in_w + (long long)(512 + kv * 512) * D_MODEL;
      p.bs1 = 0; p.bs2 = 0; p.b_ld = D_MODEL;
      p.bias = acc_in_b + 512 + kv * 512; p.res = nullptr;
      p.out = kv ? vb_ : kb_; p.os1 = ES; p.os2 = 0; p.o_ld = D_MODEL;
      p.K = D_MODEL; p.gdiv = 1;
      GEMM_XW_BIAS_F16OUT<<<dim3(BATCH, 2, 4), 256, 0, stream>>>(p);
    }
    // scores = q' k'^T
    {
      GemmP p{};
      p.X = qb; p.xs1 = ES; p.xs2 = 0; p.x_ld = D_MODEL;
      p.B = kb_; p.bs1 = ES; p.bs2 = 0; p.b_ld = D_MODEL;
      p.bias = nullptr; p.res = nullptr;
      p.out = sbuf; p.os1 = (long long)KERN * KERN; p.os2 = 0; p.o_ld = KERN;
      p.K = D_MODEL; p.gdiv = 1;
      GEMM_SCORES<<<dim3(BATCH, 2, 2), 256, 0, stream>>>(p);
    }
    // softmax (scale folded in)
    softmax256_kernel<<<(BATCH * KERN) / 8, 256, 0, stream>>>(sbuf, pbuf, scan_scale);
    // o = P @ v'
    {
      GemmP p{};
      p.X = pbuf; p.xs1 = (long long)KERN * KERN; p.xs2 = 0; p.x_ld = KERN;
      p.B = vb_; p.bs1 = ES; p.bs2 = 0; p.b_ld = D_MODEL;
      p.bias = nullptr; p.res = nullptr;
      p.out = obuf2; p.os1 = ES; p.os2 = 0; p.o_ld = D_MODEL;
      p.K = KERN; p.gdiv = 1;
      GEMM_PV<<<dim3(BATCH, 2, 4), 256, 0, stream>>>(p);
    }
    // cum_next = o @ Wo^T + bo + xs[:,wi]
    {
      float* outp = (s == 13) ? (float*)d_out : ((s & 1) ? cumB : cumA);
      GemmP p{};
      p.X = obuf2; p.xs1 = ES; p.xs2 = 0; p.x_ld = D_MODEL;
      p.B = acc_out_w; p.bs1 = 0; p.bs2 = 0; p.b_ld = D_MODEL;
      p.bias = acc_out_b;
      p.res = xs + (long long)wi * ES; p.rs1 = (long long)NWIN * ES; p.rs2 = 0; p.r_ld = D_MODEL;
      p.out = outp; p.os1 = ES; p.os2 = 0; p.o_ld = D_MODEL;
      p.K = D_MODEL; p.gdiv = 1;
      GEMM_XW_BIAS_RES<<<dim3(BATCH, 2, 4), 256, 0, stream>>>(p);
      cumprev = outp;
      cum_s1 = ES;
    }
  }
}
